// LinearAttention_17798344475097
// MI455X (gfx1250) — compile-verified, matrix operand order checked by reading, not run on hardware
//
#include <hip/hip_runtime.h>
#include <math.h>

// ---------------------------------------------------------------------------
// LinearAttention for MI455X (gfx1250): bf16 WMMA everywhere, fp32 accum.
// B=16, C=256, H=W=64 (N=4096), HEADS=8, DIM_HEAD=64, HIDDEN=512.
// GEMM inner loops are software-pipelined: next K-step's fragments are
// loaded before the current K-step's WMMAs so v_wmma never waits on a
// just-issued global_load.
// ---------------------------------------------------------------------------

typedef __bf16 bf16_t;
typedef __attribute__((ext_vector_type(16))) __bf16 v16bf;
typedef __attribute__((ext_vector_type(8)))  float  v8f;
typedef __attribute__((ext_vector_type(4)))  unsigned int uint4v;

#define NB  16
#define NC  256
#define NN  4096
#define NH  8
#define ND  64
#define NHID 512
#define NQKV 1536

union FragU { uint4v u4[2]; v16bf v; };

// 16-element bf16 WMMA operand fragment (wave32): lane = (row&15, khalf);
// elements 0..7 hold K = kk + khalf*8 + {0..7}, elements 8..15 hold K+16.
// Both runs are 16B-contiguous -> two b128 loads.
__device__ __forceinline__ v16bf ldfrag(const bf16_t* p) {
    FragU f;
    f.u4[0] = *reinterpret_cast<const uint4v*>(p);
    f.u4[1] = *reinterpret_cast<const uint4v*>(p + 16);
    return f.v;
}

__device__ __forceinline__ v8f wmma_bf16(v16bf a, v16bf b, v8f c) {
    return __builtin_amdgcn_wmma_f32_16x16x32_bf16(
        /*neg_a=*/false, a, /*neg_b=*/false, b,
        /*c_mod=*/(short)0, c, /*reuse_a=*/false, /*reuse_b=*/false);
}

// ---------------------------------------------------------------------------
// K0: fold g_norm * sqrt(C) into w_qkv (bf16), convert w_out to bf16
// ---------------------------------------------------------------------------
__global__ void k_fold_weights(const float* __restrict__ wqkv,
                               const float* __restrict__ gnorm,
                               const float* __restrict__ wout,
                               bf16_t* __restrict__ Wg,
                               bf16_t* __restrict__ Wo) {
    const int N1 = NQKV * NC;
    const int N2 = NC * NHID;
    int gid = blockIdx.x * blockDim.x + threadIdx.x;
    if (gid < N1) {
        int c = gid & (NC - 1);
        Wg[gid] = (bf16_t)(wqkv[gid] * gnorm[c] * 16.0f);   // sqrt(256) = 16
    } else if (gid < N1 + N2) {
        int i = gid - N1;
        Wo[i] = (bf16_t)wout[i];
    }
}

// ---------------------------------------------------------------------------
// K1: per-pixel inverse channel L2 norm + transpose x -> xT[b][n][c] (bf16)
// ---------------------------------------------------------------------------
__global__ void k_norm_transpose(const float* __restrict__ x,
                                 bf16_t* __restrict__ xT,
                                 float* __restrict__ invn) {
    __shared__ bf16_t tile[64][NC + 8];
    __shared__ float  sp[256];
    const int t = threadIdx.x;
    const int nl = t & 63, cq = t >> 6;           // 64 pixels x 4 channel groups
    const int b = blockIdx.y;
    const int nbase = blockIdx.x * 64;
    const float* xb = x + (size_t)b * NC * NN;
    float ss = 0.f;
    for (int c = cq; c < NC; c += 4) {
        float v = xb[(size_t)c * NN + nbase + nl];
        ss += v * v;
        tile[nl][c] = (bf16_t)v;
    }
    sp[cq * 64 + nl] = ss;
    __syncthreads();
    if (t < 64) {
        float s = sp[t] + sp[64 + t] + sp[128 + t] + sp[192 + t];
        invn[(size_t)b * NN + nbase + t] = 1.0f / fmaxf(sqrtf(s), 1e-12f);
    }
    bf16_t* dst = xT + ((size_t)b * NN + nbase) * NC;
    for (int r = 0; r < 64; ++r)
        dst[(size_t)r * NC + t] = tile[r][t];
}

// ---------------------------------------------------------------------------
// K2: QKV GEMM  qkv[b][o][n] = invn[b][n] * sum_c Wg[o][c] * xT[b][n][c]
//     8 waves/block; wave computes 32(o) x 64(n): 2 A-frags + 4 B-frags per
//     K-step feed 8 WMMAs (0.75 loads/WMMA), double-buffered over K.
// ---------------------------------------------------------------------------
__global__ void k_qkv_gemm(const bf16_t* __restrict__ Wg,
                           const bf16_t* __restrict__ xT,
                           const float* __restrict__ invn,
                           bf16_t* __restrict__ qkv) {
    const int lane = threadIdx.x & 31, wv = threadIdx.x >> 5;
    const int b = blockIdx.z;
    const int obase = blockIdx.y * 256 + wv * 32;
    const int nbase = blockIdx.x * 64;
    const int khalf = (lane >> 4) << 3;
    const int l15 = lane & 15;

    const bf16_t* arow0 = Wg + (size_t)(obase + l15) * NC + khalf;
    const bf16_t* arow1 = arow0 + (size_t)16 * NC;
    const bf16_t* xb    = xT + (size_t)b * NN * NC;
    const bf16_t* brow[4];
#pragma unroll
    for (int t = 0; t < 4; ++t)
        brow[t] = xb + (size_t)(nbase + t * 16 + l15) * NC + khalf;

    v8f acc[8] = {};                              // [o-tile 0/1][n-tile 0..3]
    v16bf a0 = ldfrag(arow0), a1 = ldfrag(arow1);
    v16bf bf[4];
#pragma unroll
    for (int t = 0; t < 4; ++t) bf[t] = ldfrag(brow[t]);

    for (int kk = 32; kk < NC; kk += 32) {
        // issue next K-step's loads first (hidden behind the 8 WMMAs below)
        __builtin_prefetch(arow0 + kk + 64, 0, 1);
        v16bf na0 = ldfrag(arow0 + kk), na1 = ldfrag(arow1 + kk);
        v16bf nb[4];
#pragma unroll
        for (int t = 0; t < 4; ++t) nb[t] = ldfrag(brow[t] + kk);
#pragma unroll
        for (int t = 0; t < 4; ++t) {
            acc[t]     = wmma_bf16(a0, bf[t], acc[t]);
            acc[4 + t] = wmma_bf16(a1, bf[t], acc[4 + t]);
        }
        a0 = na0; a1 = na1;
#pragma unroll
        for (int t = 0; t < 4; ++t) bf[t] = nb[t];
    }
#pragma unroll
    for (int t = 0; t < 4; ++t) {
        acc[t]     = wmma_bf16(a0, bf[t], acc[t]);
        acc[4 + t] = wmma_bf16(a1, bf[t], acc[4 + t]);
    }

    // epilogue: scale by invn[n], write bf16
    const int mhi = (lane >> 4) * 8;
    const float* invb = invn + (size_t)b * NN;
    bf16_t* outb = qkv + (size_t)b * NQKV * NN;
#pragma unroll
    for (int t = 0; t < 4; ++t) {
        int n = nbase + t * 16 + l15;
        float sc = invb[n];
#pragma unroll
        for (int j = 0; j < 8; ++j) {
            int o = obase + mhi + j;
            outb[(size_t)o * NN + n]        = (bf16_t)(acc[t][j] * sc);
            outb[(size_t)(o + 16) * NN + n] = (bf16_t)(acc[4 + t][j] * sc);
        }
    }
}

// ---------------------------------------------------------------------------
// K3: softmax over d (=64) per (b,h,n); writes qT[b*8+h][n][d], SCALE folded
// ---------------------------------------------------------------------------
__global__ void k_softmax_q(const bf16_t* __restrict__ qkv,
                            bf16_t* __restrict__ qT) {
    int gid = blockIdx.x * blockDim.x + threadIdx.x;
    if (gid >= NB * NH * NN) return;
    int n = gid & (NN - 1);
    int h = (gid >> 12) & 7;
    int b = gid >> 15;
    const bf16_t* src = qkv + ((size_t)b * NQKV + h * ND) * NN + n;
    float m = -1e30f;
    for (int d = 0; d < ND; ++d) m = fmaxf(m, (float)src[(size_t)d * NN]);
    float s = 0.f;
    for (int d = 0; d < ND; ++d) s += __expf((float)src[(size_t)d * NN] - m);
    float inv = 0.125f / s;                      // SCALE = 64^-0.5 = 0.125
    bf16_t* dst = qT + ((size_t)(b * NH + h) * NN + n) * ND;
    for (int d = 0; d < ND; ++d)
        dst[d] = (bf16_t)(__expf((float)src[(size_t)d * NN] - m) * inv);
}

// ---------------------------------------------------------------------------
// K4: softmax over n (=4096) per (b,h,d) row; block-per-row
// ---------------------------------------------------------------------------
__global__ void k_softmax_k(const bf16_t* __restrict__ qkv,
                            bf16_t* __restrict__ ks) {
    __shared__ float red[8];
    const int row = blockIdx.x;                   // b*512 + h*64 + d
    const int d = row & 63, h = (row >> 6) & 7, b = row >> 9;
    const bf16_t* src = qkv + ((size_t)b * NQKV + NHID + h * ND + d) * NN;
    bf16_t* dst = ks + (size_t)row * NN;
    const int t = threadIdx.x;

    float m = -1e30f;
    for (int i = t; i < NN; i += 256) m = fmaxf(m, (float)src[i]);
#pragma unroll
    for (int off = 16; off; off >>= 1) m = fmaxf(m, __shfl_down(m, off, 32));
    if ((t & 31) == 0) red[t >> 5] = m;
    __syncthreads();
    float mall = red[0];
#pragma unroll
    for (int i = 1; i < 8; ++i) mall = fmaxf(mall, red[i]);
    __syncthreads();

    float s = 0.f;
    for (int i = t; i < NN; i += 256) s += __expf((float)src[i] - mall);
#pragma unroll
    for (int off = 16; off; off >>= 1) s += __shfl_down(s, off, 32);
    if ((t & 31) == 0) red[t >> 5] = s;
    __syncthreads();
    float sall = 0.f;
#pragma unroll
    for (int i = 0; i < 8; ++i) sall += red[i];
    float inv = 1.0f / sall;
    for (int i = t; i < NN; i += 256)
        dst[i] = (bf16_t)(__expf((float)src[i] - mall) * inv);
}

// ---------------------------------------------------------------------------
// K5: context[d][e] = sum_n ks[d][n] * v[e][n]; K = 4096, double-buffered;
//     stored transposed (ctxT[e][d]) = B operand of the next GEMM.
// ---------------------------------------------------------------------------
__global__ void k_context(const bf16_t* __restrict__ ks,
                          const bf16_t* __restrict__ qkv,
                          bf16_t* __restrict__ ctxT) {
    const int bh = blockIdx.x;                    // 0..127
    const int h = bh & 7, b = bh >> 3;
    const int lane = threadIdx.x & 31, wv = threadIdx.x >> 5;
    const int dt = wv >> 1;                       // d-tile 0..3
    const int et0 = (wv & 1) * 2;                 // first of 2 e-tiles
    const int khalf = (lane >> 4) << 3;
    const int l15 = lane & 15;

    const bf16_t* arow  = ks + ((size_t)bh * ND + dt * 16 + l15) * NN + khalf;
    const bf16_t* vbase = qkv + ((size_t)b * NQKV + 2 * NHID + h * ND) * NN;
    const bf16_t* brow0 = vbase + (size_t)(et0 * 16 + l15) * NN + khalf;
    const bf16_t* brow1 = brow0 + (size_t)16 * NN;

    v8f acc0 = {}, acc1 = {};
    v16bf a = ldfrag(arow), b0 = ldfrag(brow0), b1 = ldfrag(brow1);
    for (int n = 32; n < NN; n += 32) {
        __builtin_prefetch(arow + n + 96, 0, 1);
        v16bf na = ldfrag(arow + n);
        v16bf nb0 = ldfrag(brow0 + n), nb1 = ldfrag(brow1 + n);
        acc0 = wmma_bf16(a, b0, acc0);
        acc1 = wmma_bf16(a, b1, acc1);
        a = na; b0 = nb0; b1 = nb1;
    }
    acc0 = wmma_bf16(a, b0, acc0);
    acc1 = wmma_bf16(a, b1, acc1);

    bf16_t* cb = ctxT + (size_t)bh * ND * ND;
    const int mhi = (lane >> 4) * 8;
#pragma unroll
    for (int j = 0; j < 8; ++j) {
        int dd = dt * 16 + mhi + j;
        int e0 = et0 * 16 + l15;
        cb[(size_t)e0 * ND + dd]        = (bf16_t)acc0[j];
        cb[(size_t)(e0 + 16) * ND + dd] = (bf16_t)acc1[j];
    }
}

// ---------------------------------------------------------------------------
// K6: attnT[b][n][h*64+e] = sum_d qT[bh][n][d] * ctxT[bh][e][d]; K = 64
//     (all 10 fragments loaded up front, then 8 WMMAs)
// ---------------------------------------------------------------------------
__global__ void k_attn_out(const bf16_t* __restrict__ qT,
                           const bf16_t* __restrict__ ctxT,
                           bf16_t* __restrict__ attnT) {
    const int b = blockIdx.z, h = blockIdx.y;
    const int lane = threadIdx.x & 31, wv = threadIdx.x >> 5;
    const int nbase = blockIdx.x * 128 + wv * 16;
    const int khalf = (lane >> 4) << 3;
    const int l15 = lane & 15;
    const int bh = b * NH + h;

    const bf16_t* arow = qT + ((size_t)bh * NN + nbase + l15) * ND + khalf;
    const bf16_t* cb   = ctxT + (size_t)bh * ND * ND;

    v16bf a0 = ldfrag(arow), a1 = ldfrag(arow + 32);
    v16bf bfr[4][2];
#pragma unroll
    for (int t = 0; t < 4; ++t) {
        const bf16_t* brow = cb + (size_t)(t * 16 + l15) * ND + khalf;
        bfr[t][0] = ldfrag(brow);
        bfr[t][1] = ldfrag(brow + 32);
    }
    v8f acc[4] = {};
#pragma unroll
    for (int t = 0; t < 4; ++t) {
        acc[t] = wmma_bf16(a0, bfr[t][0], acc[t]);
        acc[t] = wmma_bf16(a1, bfr[t][1], acc[t]);
    }
    bf16_t* dst = attnT + ((size_t)b * NN + nbase) * NHID + h * ND;
    const int mhi = (lane >> 4) * 8;
#pragma unroll
    for (int t = 0; t < 4; ++t) {
        int e = t * 16 + l15;
#pragma unroll
        for (int j = 0; j < 8; ++j)
            dst[(size_t)(mhi + j) * NHID + e] = (bf16_t)acc[t][j];
    }
}

// ---------------------------------------------------------------------------
// K7: projection GEMM (K=512, double-buffered) with fused bias + RMSNorm
//     epilogue; block covers all 256 channels x 32 pixels.
// ---------------------------------------------------------------------------
__global__ void k_proj_norm(const bf16_t* __restrict__ Wo,
                            const bf16_t* __restrict__ attnT,
                            const float* __restrict__ bout,
                            const float* __restrict__ gout,
                            float* __restrict__ out) {
    __shared__ float ssq[32];
    const int b = blockIdx.y;
    const int nbase = blockIdx.x * 32;
    const int t = threadIdx.x, lane = t & 31, wv = t >> 5;
    const int khalf = (lane >> 4) << 3;
    const int l15 = lane & 15;
    const int o0 = wv * 32;                       // wave owns 32 channels

    const bf16_t* arow0 = Wo + (size_t)(o0 + l15) * NHID + khalf;
    const bf16_t* arow1 = arow0 + (size_t)16 * NHID;
    const bf16_t* bb    = attnT + (size_t)b * NN * NHID;
    const bf16_t* brow0 = bb + (size_t)(nbase + l15) * NHID + khalf;
    const bf16_t* brow1 = brow0 + (size_t)16 * NHID;

    v8f acc[4] = {};                              // [o-tile][n-tile]
    v16bf a0 = ldfrag(arow0), a1 = ldfrag(arow1);
    v16bf b0 = ldfrag(brow0), b1 = ldfrag(brow1);
    for (int kk = 32; kk < NHID; kk += 32) {
        __builtin_prefetch(arow0 + kk + 64, 0, 1);
        v16bf na0 = ldfrag(arow0 + kk), na1 = ldfrag(arow1 + kk);
        v16bf nb0 = ldfrag(brow0 + kk), nb1 = ldfrag(brow1 + kk);
        acc[0] = wmma_bf16(a0, b0, acc[0]);
        acc[1] = wmma_bf16(a0, b1, acc[1]);
        acc[2] = wmma_bf16(a1, b0, acc[2]);
        acc[3] = wmma_bf16(a1, b1, acc[3]);
        a0 = na0; a1 = na1; b0 = nb0; b1 = nb1;
    }
    acc[0] = wmma_bf16(a0, b0, acc[0]);
    acc[1] = wmma_bf16(a0, b1, acc[1]);
    acc[2] = wmma_bf16(a1, b0, acc[2]);
    acc[3] = wmma_bf16(a1, b1, acc[3]);

    if (t < 32) ssq[t] = 0.f;
    __syncthreads();

    const int mhi = (lane >> 4) * 8;
    float vals[4][8];
#pragma unroll
    for (int q = 0; q < 4; ++q) {
        int ot = q >> 1, nt = q & 1;
#pragma unroll
        for (int j = 0; j < 8; ++j) {
            int o = o0 + ot * 16 + mhi + j;
            float v = acc[q][j] + bout[o];
            vals[q][j] = v;
            atomicAdd(&ssq[nt * 16 + l15], v * v);   // ds_add_f32
        }
    }
    __syncthreads();

#pragma unroll
    for (int q = 0; q < 4; ++q) {
        int ot = q >> 1, nt = q & 1;
        int nl = nt * 16 + l15;
        float sc = 16.0f / fmaxf(sqrtf(ssq[nl]), 1e-12f);  // sqrt(C)/||.||
#pragma unroll
        for (int j = 0; j < 8; ++j) {
            int o = o0 + ot * 16 + mhi + j;
            out[((size_t)b * NC + o) * NN + nbase + nl] = vals[q][j] * sc * gout[o];
        }
    }
}

// ---------------------------------------------------------------------------
// Host launcher
// ---------------------------------------------------------------------------
extern "C" void kernel_launch(void* const* d_in, const int* in_sizes, int n_in,
                              void* d_out, int out_size, void* d_ws, size_t ws_size,
                              hipStream_t stream) {
    const float* x     = (const float*)d_in[0];
    const float* gnorm = (const float*)d_in[1];
    const float* wqkv  = (const float*)d_in[2];
    const float* wout  = (const float*)d_in[3];
    const float* bout  = (const float*)d_in[4];
    const float* gout  = (const float*)d_in[5];
    float* out = (float*)d_out;

    char* ws = (char*)d_ws;
    size_t off = 0;
    auto take = [&](size_t bytes) { char* p = ws + off; off += (bytes + 255) & ~size_t(255); return p; };

    bf16_t* Wg    = (bf16_t*)take((size_t)NQKV * NC * 2);          // 768 KB
    bf16_t* Wo    = (bf16_t*)take((size_t)NC * NHID * 2);          // 256 KB
    float*  invn  = (float*) take((size_t)NB * NN * 4);            // 256 KB
    bf16_t* xT    = (bf16_t*)take((size_t)NB * NN * NC * 2);       // 32 MB
    bf16_t* qkv   = (bf16_t*)take((size_t)NB * NQKV * NN * 2);     // 192 MB
    bf16_t* qT    = (bf16_t*)take((size_t)NB * NH * NN * ND * 2);  // 64 MB
    bf16_t* ksoft = (bf16_t*)take((size_t)NB * NH * ND * NN * 2);  // 64 MB
    bf16_t* ctxT  = (bf16_t*)take((size_t)NB * NH * ND * ND * 2);  // 1 MB
    bf16_t* attnT = (bf16_t*)take((size_t)NB * NN * NHID * 2);     // 64 MB

    k_fold_weights<<<dim3((NQKV * NC + NC * NHID + 255) / 256), dim3(256), 0, stream>>>(
        wqkv, gnorm, wout, Wg, Wo);
    k_norm_transpose<<<dim3(NN / 64, NB), dim3(256), 0, stream>>>(x, xT, invn);
    k_qkv_gemm<<<dim3(NN / 64, NQKV / 256, NB), dim3(256), 0, stream>>>(Wg, xT, invn, qkv);
    k_softmax_q<<<dim3(NB * NH * NN / 256), dim3(256), 0, stream>>>(qkv, qT);
    k_softmax_k<<<dim3(NB * NH * ND), dim3(256), 0, stream>>>(qkv, ksoft);
    k_context<<<dim3(NB * NH), dim3(256), 0, stream>>>(ksoft, qkv, ctxT);
    k_attn_out<<<dim3(NN / 128, NH, NB), dim3(256), 0, stream>>>(qT, ctxT, attnT);
    k_proj_norm<<<dim3(NN / 32, NB), dim3(256), 0, stream>>>(Wo, attnT, bout, gout, out);
}
